// Head_6227702580090
// MI455X (gfx1250) — matrix operand order repaired and checked by reading, not compiled, not run on hardware
//
#include <hip/hip_runtime.h>

typedef __attribute__((ext_vector_type(16))) _Float16 v16h;
typedef __attribute__((ext_vector_type(8)))  _Float16 h8;
typedef __attribute__((ext_vector_type(4)))  _Float16 h4;
typedef __attribute__((ext_vector_type(8)))  float    v8f;
typedef __attribute__((ext_vector_type(4)))  float    f4;

constexpr int BB = 8;      // batch
constexpr int TT = 2048;   // sequence
constexpr int CC = 1024;   // embed
constexpr int HH = 64;     // head size

#define NEG_INF (-__builtin_huge_valf())

__device__ inline v8f wmma_f16(v16h a, v16h b, v8f c) {
  return __builtin_amdgcn_wmma_f32_16x16x32_f16(false, a, false, b, (short)0, c,
                                                false, false);
}

__device__ inline void cvt4(v16h& a, int base, f4 v) {
#pragma unroll
  for (int j = 0; j < 4; ++j) a[base + j] = (_Float16)v[j];
}

// ---------------------------------------------------------------------------
// Kernel 0: one-shot f32->f16 conversion of the three weight matrices into
// wh[3][64][1024].  768 KB read / 384 KB write; runs once, stays in L2.
// ---------------------------------------------------------------------------
__global__ __launch_bounds__(256) void cvt_w_kernel(
    const float* __restrict__ Wk, const float* __restrict__ Wq,
    const float* __restrict__ Wv, _Float16* __restrict__ wh) {
  const int idx = (blockIdx.x * 256 + threadIdx.x) * 4;  // over 3*64*1024
  const int which = idx / (HH * CC);
  const int off   = idx % (HH * CC);
  const float* W = (which == 0) ? Wk : (which == 1) ? Wq : Wv;
  f4 v = *(const f4*)(W + off);
  h4 h;
#pragma unroll
  for (int j = 0; j < 4; ++j) h[j] = (_Float16)v[j];
  *(h4*)(wh + idx) = h;
}

// ---------------------------------------------------------------------------
// Kernel 1: K/Q/V projections fused: each wave owns a 16-row x-tile and
// computes all 3x64 output columns (12 WMMAs per k-step from ONE A-tile
// load+convert).  B tiles are loaded in groups of 6 into distinct registers
// so loads overlap the WMMAs; next k-step's A floats are prefetched into
// registers during the current step's compute.  V is stored transposed as
// vT[B][64][T] f16 for the P*V GEMM.
// ---------------------------------------------------------------------------
__global__ __launch_bounds__(256) void proj_kernel(
    const float* __restrict__ x, const _Float16* __restrict__ wh,
    _Float16* __restrict__ kO, _Float16* __restrict__ qO,
    _Float16* __restrict__ vT) {
  const int wave = threadIdx.x >> 5;
  const int lane = threadIdx.x & 31;
  const int l16  = lane & 15;
  const int half = lane >> 4;
  const int rowA = blockIdx.x * 128 + wave * 16 + l16;  // row for A operand

  v8f acc[12] = {};

  // Software-pipelined A staging (f32, converted to f16 at use).
  const float* xa = x + (size_t)rowA * CC + half * 8;
  f4 x0 = *(const f4*)(xa + 0);
  f4 x1 = *(const f4*)(xa + 4);
  f4 x2 = *(const f4*)(xa + 16);
  f4 x3 = *(const f4*)(xa + 20);

  for (int kb = 0; kb < CC; kb += 32) {
    v16h a;
    cvt4(a, 0, x0);
    cvt4(a, 4, x1);
    cvt4(a, 8, x2);
    cvt4(a, 12, x3);
    if (kb + 32 < CC) {  // prefetch next k-step's A operand into registers
      const float* xn = xa + kb + 32;
      x0 = *(const f4*)(xn + 0);
      x1 = *(const f4*)(xn + 4);
      x2 = *(const f4*)(xn + 16);
      x3 = *(const f4*)(xn + 20);
    }
#pragma unroll
    for (int g = 0; g < 2; ++g) {
      v16h bm[6];  // distinct regs per tile -> loads overlap WMMAs
#pragma unroll
      for (int i = 0; i < 6; ++i) {
        const int wnt = g * 6 + i;  // (w*4 + nt)
        const _Float16* wb =
            wh + ((size_t)(wnt * 16 + l16)) * CC + kb + half * 8;
        *(h8*)&bm[i]       = *(const h8*)(wb);
        *((h8*)&bm[i] + 1) = *(const h8*)(wb + 16);
      }
#pragma unroll
      for (int i = 0; i < 6; ++i)
        acc[g * 6 + i] = wmma_f16(a, bm[i], acc[g * 6 + i]);
    }
  }

  // Epilogue: C layout = row (vgpr r + 8*half), col (lane%16) per 16-col tile.
#pragma unroll
  for (int w = 0; w < 3; ++w) {
#pragma unroll
    for (int nt = 0; nt < 4; ++nt) {
#pragma unroll
      for (int r = 0; r < 8; ++r) {
        const int rowg = blockIdx.x * 128 + wave * 16 + r + 8 * half;
        const int col  = nt * 16 + l16;
        const _Float16 val = (_Float16)acc[w * 4 + nt][r];
        if (w == 2) {
          const int b = rowg / TT, t = rowg % TT;
          vT[((size_t)b * HH + col) * TT + t] = val;
        } else {
          _Float16* dst = (w == 0) ? kO : qO;
          dst[(size_t)rowg * HH + col] = val;
        }
      }
    }
  }
}

// ---------------------------------------------------------------------------
// Kernel 2: flash attention.  Reference computes wei = k @ q^T (k = "query"
// role), causal tril mask, softmax over s, out = wei @ v.
// grid = (T/128, B); 8 waves/WG; each wave owns a 16-row t-tile.
// ---------------------------------------------------------------------------
__global__ __launch_bounds__(256) void attn_kernel(
    const _Float16* __restrict__ kf, const _Float16* __restrict__ qf,
    const _Float16* __restrict__ vT, float* __restrict__ out) {
  __shared__ _Float16 stg[8][16][32];  // per-wave P staging (C->A relayout)

  const int b    = blockIdx.y;
  const int wave = threadIdx.x >> 5;
  const int lane = threadIdx.x & 31;
  const int l16  = lane & 15;
  const int half = lane >> 4;
  const int t0   = blockIdx.x * 128 + wave * 16;

  // Load the 16x64 K tile (A operand), two 16x32 f16 chunks, once.
  v16h ka[2];
#pragma unroll
  for (int c = 0; c < 2; ++c) {
    const _Float16* p = kf + ((size_t)b * TT + t0 + l16) * HH + c * 32 + half * 8;
    *(h8*)&ka[c]       = *(const h8*)(p);
    *((h8*)&ka[c] + 1) = *(const h8*)(p + 16);
  }

  v8f O[4] = {};
  float m[8], l[8];
#pragma unroll
  for (int r = 0; r < 8; ++r) { m[r] = NEG_INF; l[r] = 0.0f; }

  const int send = t0 + 16;  // causal bound: need s <= t0+15
  for (int s0 = 0; s0 < send; s0 += 32) {
    // Prefetch next s-block's Q rows and V^T lines (global_prefetch_b8).
    if (s0 + 32 < send) {
      __builtin_prefetch(qf + ((size_t)b * TT + s0 + 32 + l16) * HH, 0, 0);
      __builtin_prefetch(vT + ((size_t)b * HH + l16) * TT + s0 + 32, 0, 0);
      __builtin_prefetch(vT + ((size_t)b * HH + 48 + l16) * TT + s0 + 32, 0, 0);
    }

    // ---- scores: two 16x16 tiles over columns [s0,s0+16) and [s0+16,s0+32)
    v8f S[2];
#pragma unroll
    for (int st = 0; st < 2; ++st) {
      v16h bq[2];  // load both K-chunks first, then both WMMAs
#pragma unroll
      for (int c = 0; c < 2; ++c) {
        const _Float16* qp =
            qf + ((size_t)b * TT + s0 + st * 16 + l16) * HH + c * 32 + half * 8;
        *(h8*)&bq[c]       = *(const h8*)(qp);
        *((h8*)&bq[c] + 1) = *(const h8*)(qp + 16);
      }
      v8f z = {};
      z = wmma_f16(ka[0], bq[0], z);
      z = wmma_f16(ka[1], bq[1], z);
      S[st] = z;
    }

    // ---- online softmax per row (rows live in the two 16-lane halves)
#pragma unroll
    for (int r = 0; r < 8; ++r) {
      const int rowg = t0 + r + 8 * half;
      float s0v = S[0][r] * 0.03125f;  // 1/sqrt(1024)
      float s1v = S[1][r] * 0.03125f;
      if (s0 + l16 > rowg)      s0v = NEG_INF;
      if (s0 + 16 + l16 > rowg) s1v = NEG_INF;

      float mt = fmaxf(s0v, s1v);
#pragma unroll
      for (int msk = 8; msk >= 1; msk >>= 1)
        mt = fmaxf(mt, __shfl_xor(mt, msk, 32));
      const float mn   = fmaxf(m[r], mt);
      const float corr = __expf(m[r] - mn);
      const float p0   = __expf(s0v - mn);
      const float p1   = __expf(s1v - mn);
      float rs = p0 + p1;
#pragma unroll
      for (int msk = 8; msk >= 1; msk >>= 1) rs += __shfl_xor(rs, msk, 32);
      l[r] = l[r] * corr + rs;
      m[r] = mn;
#pragma unroll
      for (int dt = 0; dt < 4; ++dt) O[dt][r] *= corr;

      stg[wave][r + 8 * half][l16]      = (_Float16)p0;
      stg[wave][r + 8 * half][16 + l16] = (_Float16)p1;
    }

    // DS is in-order per wave; fence the compiler and drain the DS counter.
    asm volatile("s_wait_dscnt 0" ::: "memory");

    // ---- reload P in the 16x32 A layout
    v16h pa;
    {
      const _Float16* p = &stg[wave][l16][half * 8];
      *(h8*)&pa       = *(const h8*)(p);
      *((h8*)&pa + 1) = *(const h8*)(p + 16);
    }

    // ---- O += P * V  (V from transposed vT: contiguous along s).
    // Load all 4 B tiles into distinct regs, then issue the 4 WMMAs.
    v16h bv[4];
#pragma unroll
    for (int dt = 0; dt < 4; ++dt) {
      const _Float16* vp =
          vT + ((size_t)b * HH + dt * 16 + l16) * TT + s0 + half * 8;
      *(h8*)&bv[dt]       = *(const h8*)(vp);
      *((h8*)&bv[dt] + 1) = *(const h8*)(vp + 16);
    }
#pragma unroll
    for (int dt = 0; dt < 4; ++dt) O[dt] = wmma_f16(pa, bv[dt], O[dt]);

    asm volatile("" ::: "memory");
  }

  // ---- epilogue: normalize and store f32 output [B,T,64]
#pragma unroll
  for (int r = 0; r < 8; ++r) {
    const int rowg = t0 + r + 8 * half;
    const float inv = 1.0f / l[r];
#pragma unroll
    for (int dt = 0; dt < 4; ++dt)
      out[((size_t)b * TT + rowg) * HH + dt * 16 + l16] = O[dt][r] * inv;
  }
}

extern "C" void kernel_launch(void* const* d_in, const int* in_sizes, int n_in,
                              void* d_out, int out_size, void* d_ws,
                              size_t ws_size, hipStream_t stream) {
  const float* x  = (const float*)d_in[0];
  const float* Wk = (const float*)d_in[1];
  const float* Wq = (const float*)d_in[2];
  const float* Wv = (const float*)d_in[3];
  float* out = (float*)d_out;

  _Float16* kf = (_Float16*)d_ws;                    // [B*T, 64] f16
  _Float16* qf = kf + (size_t)BB * TT * HH;          // [B*T, 64] f16
  _Float16* vT = qf + (size_t)BB * TT * HH;          // [B, 64, T] f16
  _Float16* wh = vT + (size_t)BB * TT * HH;          // [3, 64, 1024] f16

  dim3 blk(256);
  cvt_w_kernel<<<dim3((3 * HH * CC) / (256 * 4)), blk, 0, stream>>>(Wk, Wq, Wv, wh);

  proj_kernel<<<dim3((BB * TT) / 128), blk, 0, stream>>>(x, wh, kf, qf, vT);

  attn_kernel<<<dim3(TT / 128, BB), blk, 0, stream>>>(kf, qf, vT, out);
}
